// Generator_43662637531171
// MI455X (gfx1250) — matrix operand order repaired and checked by reading, not compiled, not run on hardware
//
#include <hip/hip_runtime.h>
#include <hip/hip_bf16.h>

typedef __attribute__((ext_vector_type(16))) _Float16 v16h;
typedef __attribute__((ext_vector_type(8)))  float    v8f;

union HFrag { v16h h; uint4 q[2]; };

#define GNN_BN   4096      // B*N nodes
#define GNN_H    256       // hidden dim
#define GNN_E    16384     // edges
#define GNN_SEL  65536     // selected pairs

// One K-step (K += 32): A fragment from a single segment row (aSeg + lo),
// B fragments from 4 transposed-weight rows at global offset kB.
__device__ __forceinline__ void wmma_kstep(
    const _Float16* __restrict__ aSeg, int lo, int kB, int g,
    const _Float16* __restrict__ bp0, const _Float16* __restrict__ bp1,
    const _Float16* __restrict__ bp2, const _Float16* __restrict__ bp3,
    v8f& acc0, v8f& acc1, v8f& acc2, v8f& acc3)
{
  HFrag a;
  a.q[0] = *reinterpret_cast<const uint4*>(aSeg + lo + 16 * g);       // halves 0..7
  a.q[1] = *reinterpret_cast<const uint4*>(aSeg + lo + 16 * g + 8);   // halves 8..15
  HFrag b0, b1, b2, b3;
  b0.q[0] = *reinterpret_cast<const uint4*>(bp0 + kB);
  b0.q[1] = *reinterpret_cast<const uint4*>(bp0 + kB + 8);
  b1.q[0] = *reinterpret_cast<const uint4*>(bp1 + kB);
  b1.q[1] = *reinterpret_cast<const uint4*>(bp1 + kB + 8);
  b2.q[0] = *reinterpret_cast<const uint4*>(bp2 + kB);
  b2.q[1] = *reinterpret_cast<const uint4*>(bp2 + kB + 8);
  b3.q[0] = *reinterpret_cast<const uint4*>(bp3 + kB);
  b3.q[1] = *reinterpret_cast<const uint4*>(bp3 + kB + 8);
  acc0 = __builtin_amdgcn_wmma_f32_16x16x32_f16(false, a.h, false, b0.h, (short)0, acc0, false, false);
  acc1 = __builtin_amdgcn_wmma_f32_16x16x32_f16(false, a.h, false, b1.h, (short)0, acc1, false, false);
  acc2 = __builtin_amdgcn_wmma_f32_16x16x32_f16(false, a.h, false, b2.h, (short)0, acc2, false, false);
  acc3 = __builtin_amdgcn_wmma_f32_16x16x32_f16(false, a.h, false, b3.h, (short)0, acc3, false, false);
}

// ---------------------------------------------------------------------------
// Generic WMMA GEMM: out[M x Ntot] = epi( A[M x K] * B[K x Ntot] )
//   B given transposed+f16: BT[Ntot x K]
//   A operand built per-lane by mode:
//     0: plain        A1 + m*K
//     1: gather       A1 + g1[m]*K                      (h[src] for messages)
//     2: concat2      k<256: A1+m*256   else A2+m*256   ([h | agg])
//     3: concat-prop  k<256: A1+m*256   else A2+(m>>6)*64  ([h | prop(b)])
//     4: pair-gather  k<256: A1+(sb*64+si)*256 else A1+(sb*64+sj)*256
//   epilogue mode:
//     0: relu(acc + bias[n]) -> f16 store
//     1:      acc + bias[n]  -> f16 store
//     2: relu(acc + eadd[m*256+n]) -> atomicAdd(agg[dst[m]*256+n])  (messages)
// Block: 256 threads = 8 waves; wave (wm,wn) computes 16x64; block = 32x256.
// grid = (Ntot/256, M/32).  K1 and K are multiples of 32, so the K loop is
// split into two phases (one per A segment) -> no selects in the hot loop.
// ---------------------------------------------------------------------------
__global__ __launch_bounds__(256) void k_wmma_gemm(
    const _Float16* __restrict__ A1, const _Float16* __restrict__ A2,
    int K1, int K,
    const _Float16* __restrict__ BT, int Ntot,
    const float* __restrict__ bias,
    _Float16* __restrict__ outH,
    const float* __restrict__ eadd, const int* __restrict__ dstIdx,
    float* __restrict__ agg,
    const int* __restrict__ g1,
    const int* __restrict__ gsb, const int* __restrict__ gsi,
    const int* __restrict__ gsj,
    int aMode, int epiMode)
{
  const int tid  = threadIdx.x;
  const int lane = tid & 31;
  const int wid  = tid >> 5;
  const int wm   = wid & 1;
  const int wn   = wid >> 1;
  const int g    = lane >> 4;   // lane group (0: lanes 0-15, 1: lanes 16-31)
  const int l15  = lane & 15;

  const int tileM = blockIdx.y * 32  + wm * 16;
  const int tileN = blockIdx.x * 256 + wn * 64;

  // ---- per-lane A row pointers (lane holds matrix row tileM + l15) ----
  const int mA = tileM + l15;
  const _Float16* r1 = nullptr;
  const _Float16* r2 = nullptr;
  switch (aMode) {
    case 0: r1 = A1 + (size_t)mA * K; break;
    case 1: r1 = A1 + (size_t)g1[mA] * K; break;
    case 2: r1 = A1 + (size_t)mA * 256; r2 = A2 + (size_t)mA * 256; break;
    case 3: r1 = A1 + (size_t)mA * 256; r2 = A2 + (size_t)(mA >> 6) * 64; break;
    default: {
      int base = gsb[mA] * 64;
      r1 = A1 + (size_t)(base + gsi[mA]) * 256;
      r2 = A1 + (size_t)(base + gsj[mA]) * 256;
    } break;
  }

  // ---- B row pointers: lane reads column tileN + f*16 + l15 of BT[N x K] ----
  const _Float16* bp0 = BT + (size_t)(tileN +  0 + l15) * K;
  const _Float16* bp1 = BT + (size_t)(tileN + 16 + l15) * K;
  const _Float16* bp2 = BT + (size_t)(tileN + 32 + l15) * K;
  const _Float16* bp3 = BT + (size_t)(tileN + 48 + l15) * K;

  v8f acc0 = {}; v8f acc1 = {}; v8f acc2 = {}; v8f acc3 = {};

  const int kSplit = (K1 < K) ? K1 : K;
  // phase 1: A from segment 1
  for (int kk = 0; kk < kSplit; kk += 32)
    wmma_kstep(r1, kk, kk + 16 * g, g, bp0, bp1, bp2, bp3, acc0, acc1, acc2, acc3);
  // phase 2: A from segment 2 (empty when K1 == K)
  for (int kk = kSplit; kk < K; kk += 32)
    wmma_kstep(r2, kk - K1, kk + 16 * g, g, bp0, bp1, bp2, bp3, acc0, acc1, acc2, acc3);

  // ---- epilogue: C layout -> element (m = tileM + r + 8g, n = tileN + f*16 + l15) ----
  v8f accs[4] = {acc0, acc1, acc2, acc3};
  int dstRow[8];
  if (epiMode == 2) {
    #pragma unroll
    for (int r = 0; r < 8; ++r) dstRow[r] = dstIdx[tileM + r + 8 * g];
  }
  #pragma unroll
  for (int f = 0; f < 4; ++f) {
    const int n = tileN + f * 16 + l15;
    const float bv = (epiMode == 2) ? 0.0f : bias[n];
    #pragma unroll
    for (int r = 0; r < 8; ++r) {
      const int m = tileM + r + 8 * g;
      float v = accs[f][r];
      if (epiMode == 2) {
        v += eadd[(size_t)m * 256 + n];
        v = fmaxf(v, 0.0f);
        atomicAdd(agg + (size_t)dstRow[r] * 256 + n, v);
      } else {
        v += bv;
        if (epiMode == 0) v = fmaxf(v, 0.0f);
        outH[(size_t)m * Ntot + n] = (_Float16)v;
      }
    }
  }
}

// ---------------------------------------------------------------------------
// Prep / small kernels
// ---------------------------------------------------------------------------
__global__ void k_cvt_f16(const float* __restrict__ s, _Float16* __restrict__ d, int n) {
  int i = blockIdx.x * blockDim.x + threadIdx.x;
  if (i < n) d[i] = (_Float16)s[i];
}

// W[K x N] f32 -> WT[N x K] f16
__global__ void k_transpose_cvt(const float* __restrict__ W, _Float16* __restrict__ WT,
                                int K, int N) {
  int i = blockIdx.x * blockDim.x + threadIdx.x;
  if (i < K * N) {
    int k = i / N, n = i - k * N;
    WT[(size_t)n * K + k] = (_Float16)W[i];
  }
}

__global__ void k_zero_f32(float* __restrict__ p, int n) {
  int i = blockIdx.x * blockDim.x + threadIdx.x;
  if (i < n) p[i] = 0.0f;
}

// EATTR[e,n] = b_msg[n] + sum_{j<5} edge_attr[e,j] * W_msg[256+j, n]
__global__ void k_eattr(const float* __restrict__ ea, const float* __restrict__ Wm,
                        const float* __restrict__ bm, float* __restrict__ out) {
  int i = blockIdx.x * blockDim.x + threadIdx.x;   // GNN_E * 256
  int e = i >> 8, n = i & 255;
  float v = bm[n];
  #pragma unroll
  for (int j = 0; j < 5; ++j) v += ea[e * 5 + j] * Wm[(256 + j) * 256 + n];
  out[i] = v;
}

// prop[b,p] = props[b]*W_prop[0,p] + b_prop[p]   (64 x 64, f16)
__global__ void k_prop(const float* __restrict__ props, const float* __restrict__ Wp,
                       const float* __restrict__ bp, _Float16* __restrict__ out) {
  int i = blockIdx.x * blockDim.x + threadIdx.x;   // 4096
  int b = i >> 6, p = i & 63;
  out[i] = (_Float16)(props[b] * Wp[p] + bp[p]);
}

__global__ void k_zero1(float* p) { p[0] = 0.0f; }

// Fused logits (H3[E x 512] @ W_out[512 x 5] + b_out) + log-softmax CE mean
__global__ __launch_bounds__(256) void k_loss(const _Float16* __restrict__ H3,
    const float* __restrict__ Wout, const float* __restrict__ bout,
    const int* __restrict__ gold, float* __restrict__ out)
{
  const int e = blockIdx.x * 256 + threadIdx.x;
  float lg[5];
  #pragma unroll
  for (int c = 0; c < 5; ++c) lg[c] = bout[c];
  const _Float16* row = H3 + (size_t)e * 512;
  for (int k0 = 0; k0 < 512; k0 += 8) {
    uint4 q = *reinterpret_cast<const uint4*>(row + k0);
    const _Float16* hv = reinterpret_cast<const _Float16*>(&q);
    #pragma unroll
    for (int j = 0; j < 8; ++j) {
      float x = (float)hv[j];
      #pragma unroll
      for (int c = 0; c < 5; ++c) lg[c] += x * Wout[(k0 + j) * 5 + c];
    }
  }
  float mx = lg[0];
  #pragma unroll
  for (int c = 1; c < 5; ++c) mx = fmaxf(mx, lg[c]);
  float s = 0.0f;
  #pragma unroll
  for (int c = 0; c < 5; ++c) s += __expf(lg[c] - mx);
  const int gd = gold[e];
  float lp = lg[gd] - mx - __logf(s);
  float v = -lp * (1.0f / (float)GNN_SEL);
  #pragma unroll
  for (int off = 16; off > 0; off >>= 1) v += __shfl_down(v, off, 32);
  __shared__ float sred[8];
  const int lane = threadIdx.x & 31, wid = threadIdx.x >> 5;
  if (lane == 0) sred[wid] = v;
  __syncthreads();
  if (threadIdx.x == 0) {
    float t = 0.0f;
    #pragma unroll
    for (int i = 0; i < 8; ++i) t += sred[i];
    atomicAdd(out, t);
  }
}

// ---------------------------------------------------------------------------
extern "C" void kernel_launch(void* const* d_in, const int* in_sizes, int n_in,
                              void* d_out, int out_size, void* d_ws, size_t ws_size,
                              hipStream_t stream) {
  const float* x      = (const float*)d_in[0];
  const int*   eidx   = (const int*)  d_in[1];
  const float* eattr  = (const float*)d_in[2];
  const float* props  = (const float*)d_in[3];
  const int*   sel_b  = (const int*)  d_in[4];
  const int*   sel_i  = (const int*)  d_in[5];
  const int*   sel_j  = (const int*)  d_in[6];
  const int*   golden = (const int*)  d_in[7];
  const float* W_prop = (const float*)d_in[8];
  const float* b_prop = (const float*)d_in[9];
  const float* W_in   = (const float*)d_in[10];
  const float* b_in   = (const float*)d_in[11];
  const float* W_msg  = (const float*)d_in[12];
  const float* b_msg  = (const float*)d_in[13];
  const float* W_upd  = (const float*)d_in[14];
  const float* b_upd  = (const float*)d_in[15];
  const float* W_add  = (const float*)d_in[16];
  const float* b_add  = (const float*)d_in[17];
  const float* W1     = (const float*)d_in[18];
  const float* b1     = (const float*)d_in[19];
  const float* W2     = (const float*)d_in[20];
  const float* b2     = (const float*)d_in[21];
  const float* W3     = (const float*)d_in[22];
  const float* b3     = (const float*)d_in[23];
  const float* W_out  = (const float*)d_in[24];
  const float* b_out  = (const float*)d_in[25];
  float* out = (float*)d_out;

  char* ws = (char*)d_ws;
  size_t off = 0;
  auto alloc = [&](size_t bytes) -> char* {
    char* p = ws + off;
    off = (off + bytes + 255) & ~(size_t)255;
    return p;
  };

  _Float16* Xf16   = (_Float16*)alloc((size_t)GNN_BN * 192 * 2);
  _Float16* WinT   = (_Float16*)alloc((size_t)256 * 192 * 2);
  _Float16* WmsgT  = (_Float16*)alloc((size_t)256 * 256 * 2);
  _Float16* WupdT  = (_Float16*)alloc((size_t)256 * 512 * 2);
  _Float16* WaddT  = (_Float16*)alloc((size_t)256 * 320 * 2);
  _Float16* W1T    = (_Float16*)alloc((size_t)256 * 512 * 2);
  _Float16* W2T    = (_Float16*)alloc((size_t)256 * 256 * 2);
  _Float16* W3T    = (_Float16*)alloc((size_t)512 * 256 * 2);
  _Float16* Pf16   = (_Float16*)alloc((size_t)64 * 64 * 2);
  float*    EATTR  = (float*)   alloc((size_t)GNN_E * 256 * 4);
  _Float16* HA     = (_Float16*)alloc((size_t)GNN_BN * 256 * 2);
  _Float16* HB     = (_Float16*)alloc((size_t)GNN_BN * 256 * 2);
  float*    AGG    = (float*)   alloc((size_t)GNN_BN * 256 * 4);
  _Float16* AGGh   = (_Float16*)alloc((size_t)GNN_BN * 256 * 2);
  _Float16* Gh     = (_Float16*)alloc((size_t)GNN_BN * 256 * 2);
  _Float16* H1     = (_Float16*)alloc((size_t)GNN_SEL * 256 * 2);
  _Float16* H2     = (_Float16*)alloc((size_t)GNN_SEL * 256 * 2);
  _Float16* H3     = (_Float16*)alloc((size_t)GNN_SEL * 512 * 2);

  const int* src = eidx;
  const int* dst = eidx + GNN_E;

  // ---- prep ----
  k_cvt_f16<<<(GNN_BN * 192 + 255) / 256, 256, 0, stream>>>(x, Xf16, GNN_BN * 192);
  k_transpose_cvt<<<(192 * 256 + 255) / 256, 256, 0, stream>>>(W_in,  WinT,  192, 256);
  k_transpose_cvt<<<(256 * 256 + 255) / 256, 256, 0, stream>>>(W_msg, WmsgT, 256, 256); // rows 0..255
  k_transpose_cvt<<<(512 * 256 + 255) / 256, 256, 0, stream>>>(W_upd, WupdT, 512, 256);
  k_transpose_cvt<<<(320 * 256 + 255) / 256, 256, 0, stream>>>(W_add, WaddT, 320, 256);
  k_transpose_cvt<<<(512 * 256 + 255) / 256, 256, 0, stream>>>(W1,    W1T,   512, 256);
  k_transpose_cvt<<<(256 * 256 + 255) / 256, 256, 0, stream>>>(W2,    W2T,   256, 256);
  k_transpose_cvt<<<(256 * 512 + 255) / 256, 256, 0, stream>>>(W3,    W3T,   256, 512);
  k_eattr<<<(GNN_E * 256) / 256, 256, 0, stream>>>(eattr, W_msg, b_msg, EATTR);
  k_prop<<<(64 * 64) / 256, 256, 0, stream>>>(props, W_prop, b_prop, Pf16);

  // ---- h0 = relu(X @ W_in + b_in) ----
  k_wmma_gemm<<<dim3(1, GNN_BN / 32), 256, 0, stream>>>(
      Xf16, nullptr, 192, 192, WinT, 256, b_in, HA,
      nullptr, nullptr, nullptr, nullptr, nullptr, nullptr, nullptr, 0, 0);

  // ---- 4 rounds of message passing ----
  _Float16* hc = HA;
  _Float16* hn = HB;
  for (int t = 0; t < 4; ++t) {
    k_zero_f32<<<(GNN_BN * 256) / 256, 256, 0, stream>>>(AGG, GNN_BN * 256);
    // msg = relu(h[src] @ Wmsg1 + EATTR); atomic scatter-add into AGG[dst]
    k_wmma_gemm<<<dim3(1, GNN_E / 32), 256, 0, stream>>>(
        hc, nullptr, 256, 256, WmsgT, 256, nullptr, nullptr,
        EATTR, dst, AGG, src, nullptr, nullptr, nullptr, 1, 2);
    k_cvt_f16<<<(GNN_BN * 256) / 256, 256, 0, stream>>>(AGG, AGGh, GNN_BN * 256);
    // h = relu([h | agg] @ W_upd + b_upd)
    k_wmma_gemm<<<dim3(1, GNN_BN / 32), 256, 0, stream>>>(
        hc, AGGh, 256, 512, WupdT, 256, b_upd, hn,
        nullptr, nullptr, nullptr, nullptr, nullptr, nullptr, nullptr, 2, 0);
    _Float16* tmp = hc; hc = hn; hn = tmp;
  }

  // ---- node_embg = [h | prop(b)] @ W_add + b_add (no relu) ----
  k_wmma_gemm<<<dim3(1, GNN_BN / 32), 256, 0, stream>>>(
      hc, Pf16, 256, 320, WaddT, 256, b_add, Gh,
      nullptr, nullptr, nullptr, nullptr, nullptr, nullptr, nullptr, 3, 1);

  // ---- edge predictor ----
  k_wmma_gemm<<<dim3(1, GNN_SEL / 32), 256, 0, stream>>>(
      Gh, nullptr, 256, 512, W1T, 256, b1, H1,
      nullptr, nullptr, nullptr, nullptr, sel_b, sel_i, sel_j, 4, 0);
  k_wmma_gemm<<<dim3(1, GNN_SEL / 32), 256, 0, stream>>>(
      H1, nullptr, 256, 256, W2T, 256, b2, H2,
      nullptr, nullptr, nullptr, nullptr, nullptr, nullptr, nullptr, 0, 0);
  k_wmma_gemm<<<dim3(2, GNN_SEL / 32), 256, 0, stream>>>(
      H2, nullptr, 256, 256, W3T, 512, b3, H3,
      nullptr, nullptr, nullptr, nullptr, nullptr, nullptr, nullptr, 0, 1);

  // ---- logits + cross-entropy (mean) ----
  k_zero1<<<1, 1, 0, stream>>>(out);
  k_loss<<<GNN_SEL / 256, 256, 0, stream>>>(H3, W_out, b_out, golden, out);
}
